// SigTKAN_1297080123658
// MI455X (gfx1250) — compile-verified, matrix operand order checked by reading, not run on hardware
//
#include <hip/hip_runtime.h>
#include <hip/hip_bf16.h>

// Problem constants (match reference)
#define B_   256
#define T_   512
#define F_   64
#define U_   256
#define SIG_ 4160   // F + F*F
#define NB_  8      // spline bases per input feature (G+K)

// GEMM tiling: 64x64 output tile, K-step 32, 128 threads (4 waves).
#define TM 64
#define TN 64
#define BK 32

typedef __attribute__((ext_vector_type(16))) __bf16 bf16x16;
typedef __attribute__((ext_vector_type(8)))  __bf16 bf16x8;
typedef __attribute__((ext_vector_type(8)))  float  floatx8;

#if defined(__AMDGCN__) && defined(__gfx1250__) && \
    __has_builtin(__builtin_amdgcn_tensor_load_to_lds)
#define HAVE_TDM 1
typedef __attribute__((ext_vector_type(4))) unsigned int tdm_u32x4;
typedef __attribute__((ext_vector_type(8))) int          tdm_i32x8;
typedef __attribute__((ext_vector_type(4))) int          tdm_i32x4;
#else
#define HAVE_TDM 0
#endif

__device__ __forceinline__ float sigmoidf_(float x) { return 1.0f / (1.0f + __expf(-x)); }
__device__ __forceinline__ float siluf_(float x)    { return x * sigmoidf_(x); }

// Cox-de Boor B-spline bases, K=3, G=5, knots g[i] = 0.4*i - 2.2, i=0..11.
__device__ __forceinline__ void bases8(float x, float* out) {
    float g[12];
#pragma unroll
    for (int i = 0; i < 12; ++i) g[i] = 0.4f * (float)i - 2.2f;
    float b[11];
#pragma unroll
    for (int i = 0; i < 11; ++i) b[i] = (x >= g[i] && x < g[i + 1]) ? 1.0f : 0.0f;
#pragma unroll
    for (int k = 1; k <= 3; ++k) {
#pragma unroll
        for (int i = 0; i + k < 11; ++i) {
            float d1 = g[i + k] - g[i];
            float d2 = g[i + k + 1] - g[i + 1];
            b[i] = (x - g[i]) / d1 * b[i] + (g[i + k + 1] - x) / d2 * b[i + 1];
        }
    }
#pragma unroll
    for (int q = 0; q < 8; ++q) out[q] = b[q];
}

// ---- WMMA fragment helpers (bf16 16x16x32, wave32) -------------------------
// Lane(half=lane>>4, r=lane&15) holds K in [8h,8h+8) and [16+8h,16+8h+8):
// two contiguous 16-byte spans of a row-major [16][32] bf16 tile.
__device__ __forceinline__ bf16x16 load_frag(const __bf16* tile) {
    int lane = threadIdx.x & 31;
    int half = lane >> 4, r = lane & 15;
    const __bf16* p = tile + r * BK + 8 * half;
    bf16x8 lo = *(const bf16x8*)(p);
    bf16x8 hi = *(const bf16x8*)(p + 16);
    return __builtin_shufflevector(lo, hi, 0, 1, 2, 3, 4, 5, 6, 7,
                                   8, 9, 10, 11, 12, 13, 14, 15);
}
__device__ __forceinline__ floatx8 wmma_bf16(bf16x16 a, bf16x16 b, floatx8 c) {
    return __builtin_amdgcn_wmma_f32_16x16x32_bf16(false, a, false, b, (short)0, c,
                                                   false, false);
}
// Batched tile step: load A frag + all 4 B frags, then 4 back-to-back WMMAs.
__device__ __forceinline__ void mma_step(const __bf16* As, const __bf16* Bt,
                                         int wave, floatx8* acc) {
    bf16x16 a  = load_frag(&As[(wave * 16) * BK]);
    bf16x16 b0 = load_frag(&Bt[0 * 16 * BK]);
    bf16x16 b1 = load_frag(&Bt[1 * 16 * BK]);
    bf16x16 b2 = load_frag(&Bt[2 * 16 * BK]);
    bf16x16 b3 = load_frag(&Bt[3 * 16 * BK]);
    acc[0] = wmma_bf16(a, b0, acc[0]);
    acc[1] = wmma_bf16(a, b1, acc[1]);
    acc[2] = wmma_bf16(a, b2, acc[2]);
    acc[3] = wmma_bf16(a, b3, acc[3]);
}

// Pack 8 floats -> bf16x8 -> one ds_store_b128.
__device__ __forceinline__ void store8(__bf16* dst, const float* v) {
    bf16x8 p;
#pragma unroll
    for (int q = 0; q < 8; ++q) p[q] = (__bf16)v[q];
    *(bf16x8*)dst = p;
}

// ---- Kernel 1: depth-2 path signature --------------------------------------
__global__ __launch_bounds__(256) void sig_kernel(const float* __restrict__ xin,
                                                  const float* __restrict__ tk,
                                                  float* __restrict__ sig) {
    const int CH = 64;
    __shared__ float xs[CH + 1][F_];
    int b = blockIdx.x, tid = threadIdx.x;
    int i = tid & 63, jg = tid >> 6;
    float prefix = 0.0f;
    float acc[16];
#pragma unroll
    for (int q = 0; q < 16; ++q) acc[q] = 0.0f;

    for (int t0 = 0; t0 < T_ - 1; t0 += CH) {
        int nrows = (CH + 1 < T_ - t0) ? (CH + 1) : (T_ - t0);
        for (int idx = tid; idx < nrows * F_; idx += 256) {
            int rr = idx >> 6, ff = idx & 63;
            int tg = t0 + rr;
            xs[rr][ff] = tk[tg] * xin[(b * T_ + tg) * F_ + ff];
        }
        __syncthreads();
        int steps = nrows - 1;
        for (int s = 0; s < steps; ++s) {
            float dxi = xs[s + 1][i] - xs[s][i];
            float a   = prefix + 0.5f * dxi;
#pragma unroll
            for (int q = 0; q < 16; ++q) {
                int j = jg * 16 + q;
                acc[q] += a * (xs[s + 1][j] - xs[s][j]);
            }
            prefix += dxi;
        }
        __syncthreads();
    }
    if (jg == 0) sig[b * SIG_ + i] = prefix;
#pragma unroll
    for (int q = 0; q < 16; ++q)
        sig[b * SIG_ + F_ + i * F_ + jg * 16 + q] = acc[q];
}

// ---- Kernel 2: KAN linear: out = [elu]( silu(X)@wb + bases(X)@ws ) ---------
__global__ __launch_bounds__(128) void kan_kernel(
    const float* __restrict__ X, int din, const float* __restrict__ wb,
    const float* __restrict__ ws, float* __restrict__ out, int do_elu,
    const float* __restrict__ zpre, const float* __restrict__ gateW,
    float* __restrict__ hstate, float* __restrict__ ys, int t) {
    __shared__ __bf16 As[TM * BK];   // row-major [row][32]
    __shared__ __bf16 Bt[TN * BK];   // transposed [col][32]
    int tid = threadIdx.x;
    int wave = tid >> 5, lane = tid & 31;
    int half = lane >> 4, r = lane & 15;
    int rowBlk = blockIdx.x * TM;
    int colBlk = blockIdx.y * TN;
    int arr = tid >> 1, akh = (tid & 1) * 16;
    int bc = tid & 63, bkh = (tid >> 6) * 16;

    floatx8 acc[4];
#pragma unroll
    for (int n = 0; n < 4; ++n)
#pragma unroll
        for (int e = 0; e < 8; ++e) acc[n][e] = 0.0f;

    // Phase 1: silu(X) @ wb, K = din
    for (int k0 = 0; k0 < din; k0 += BK) {
        if (tid == 0 && k0 + BK < din)
            __builtin_prefetch(&wb[(k0 + BK) * U_ + colBlk], 0, 1);
        {
            const float* src = &X[(rowBlk + arr) * din + k0 + akh];
            float v[16];
#pragma unroll
            for (int q = 0; q < 16; q += 4) {
                float4 f4 = *(const float4*)(src + q);
                v[q + 0] = siluf_(f4.x); v[q + 1] = siluf_(f4.y);
                v[q + 2] = siluf_(f4.z); v[q + 3] = siluf_(f4.w);
            }
            store8(&As[arr * BK + akh], v);
            store8(&As[arr * BK + akh + 8], v + 8);
        }
        {
            float v[16];
#pragma unroll
            for (int q = 0; q < 16; ++q)
                v[q] = wb[(k0 + bkh + q) * U_ + colBlk + bc];
            store8(&Bt[bc * BK + bkh], v);
            store8(&Bt[bc * BK + bkh + 8], v + 8);
        }
        __syncthreads();
        mma_step(As, Bt, wave, acc);
        __syncthreads();
    }

    // Phase 2: bases(X) @ ws, 4 features (=32 K) per iteration
    for (int f0 = 0; f0 < din; f0 += 4) {
        for (int idx = tid; idx < TM * 4; idx += 128) {
            int rr = idx >> 2, ff = idx & 3;
            float bb[8];
            bases8(X[(rowBlk + rr) * din + f0 + ff], bb);
            store8(&As[rr * BK + ff * 8], bb);
        }
        {
            int kbase = f0 * NB_;
            float v[16];
#pragma unroll
            for (int q = 0; q < 16; ++q)
                v[q] = ws[(kbase + bkh + q) * U_ + colBlk + bc];
            store8(&Bt[bc * BK + bkh], v);
            store8(&Bt[bc * BK + bkh + 8], v + 8);
        }
        __syncthreads();
        mma_step(As, Bt, wave, acc);
        __syncthreads();
    }

    // Epilogue
#pragma unroll
    for (int n = 0; n < 4; ++n) {
#pragma unroll
        for (int v = 0; v < 8; ++v) {
            int row = rowBlk + wave * 16 + v + 8 * half;
            int col = colBlk + n * 16 + r;
            float s = acc[n][v];
            if (zpre != nullptr) {
                float ig = sigmoidf_(zpre[row * 768 + col]);
                float og = sigmoidf_(zpre[row * 768 + 512 + col]);
                float o  = og * tanhf(ig * s) * gateW[row * U_ + col];
                hstate[row * U_ + col]        = o;
                ys[(row * T_ + t) * U_ + col] = o;
            } else {
                if (do_elu) s = (s > 0.0f) ? s : (__expf(s) - 1.0f);
                out[row * U_ + col] = s;
            }
        }
    }
}

// ---- Kernel 3: generic bf16-WMMA GEMM with TDM weight staging --------------
// out(256xN) = A(256xK, ld=lda) @ W(KxN). W tiles (32x64 fp32, row stride N)
// are DMA'd to LDS by the Tensor Data Mover, then transposed+converted.
__global__ __launch_bounds__(128) void gemm_kernel(
    const float* __restrict__ A, int lda, int K, const float* __restrict__ W,
    int N, float* __restrict__ out, int ldo) {
    __shared__ __bf16 As[TM * BK];
    __shared__ __bf16 Bt[TN * BK];
#if HAVE_TDM
    __shared__ float Wstage[BK * TN];   // 8KB fp32 staging tile
#endif
    int tid = threadIdx.x;
    int wave = tid >> 5, lane = tid & 31;
    int half = lane >> 4, r = lane & 15;
    int rowBlk = blockIdx.x * TM;
    int colBlk = blockIdx.y * TN;
    int arr = tid >> 1, akh = (tid & 1) * 16;
    int bc = tid & 63, bkh = (tid >> 6) * 16;

    floatx8 acc[4];
#pragma unroll
    for (int n = 0; n < 4; ++n)
#pragma unroll
        for (int e = 0; e < 8; ++e) acc[n][e] = 0.0f;

    for (int k0 = 0; k0 < K; k0 += BK) {
#if HAVE_TDM
        if (tid < 32) {  // wave 0 issues the tensor DMA: global W tile -> LDS
            unsigned long long ga =
                (unsigned long long)(const void*)&W[(unsigned long long)k0 * N + colBlk];
            unsigned ldsoff =
                (unsigned)(unsigned long long)(uintptr_t)(const void*)Wstage;
            unsigned rows = (unsigned)(K - k0);
            tdm_u32x4 g0;
            g0[0] = 1u;                                   // count=1
            g0[1] = ldsoff;                               // lds_addr
            g0[2] = (unsigned)ga;                         // global_addr[31:0]
            g0[3] = (unsigned)((ga >> 32) & 0x1FFFFFFull) // global_addr[56:32]
                    | (2u << 30);                         // type=2 (image)
            tdm_i32x8 g1;
            g1[0] = (int)(2u << 16);                      // data_size=4B
            g1[1] = (int)(((unsigned)N & 0xFFFFu) << 16); // tensor_dim0[15:0]
            g1[2] = (int)(((unsigned)N >> 16) | ((rows & 0xFFFFu) << 16));
            g1[3] = (int)((rows >> 16) | ((unsigned)TN << 16)); // tile_dim0=64
            g1[4] = (int)BK;                              // tile_dim1=32 rows
            g1[5] = (int)N;                               // tensor_dim0_stride
            g1[6] = 0;
            g1[7] = 0;
            tdm_i32x4 gz = {0, 0, 0, 0};
#if __clang_major__ >= 23
            tdm_i32x8 gz8 = {0, 0, 0, 0, 0, 0, 0, 0};
            __builtin_amdgcn_tensor_load_to_lds(g0, g1, gz, gz, gz8, 0);
#else
            __builtin_amdgcn_tensor_load_to_lds(g0, g1, gz, gz, 0);
#endif
            __builtin_amdgcn_s_wait_tensorcnt(0);
        }
#endif
        {   // A tile fill (overlaps with TDM)
            const float* src = &A[(rowBlk + arr) * lda + k0 + akh];
            float v[16];
#pragma unroll
            for (int q = 0; q < 16; q += 4) {
                float4 f4 = *(const float4*)(src + q);
                v[q + 0] = f4.x; v[q + 1] = f4.y; v[q + 2] = f4.z; v[q + 3] = f4.w;
            }
            store8(&As[arr * BK + akh], v);
            store8(&As[arr * BK + akh + 8], v + 8);
        }
#if HAVE_TDM
        __syncthreads();   // Wstage + As visible
        {   // transpose + convert from LDS staging
            float v[16];
#pragma unroll
            for (int q = 0; q < 16; ++q)
                v[q] = Wstage[(bkh + q) * TN + bc];
            store8(&Bt[bc * BK + bkh], v);
            store8(&Bt[bc * BK + bkh + 8], v + 8);
        }
#else
        {
            float v[16];
#pragma unroll
            for (int q = 0; q < 16; ++q)
                v[q] = W[(k0 + bkh + q) * N + colBlk + bc];
            store8(&Bt[bc * BK + bkh], v);
            store8(&Bt[bc * BK + bkh + 8], v + 8);
        }
#endif
        __syncthreads();
        mma_step(As, Bt, wave, acc);
        __syncthreads();
    }
#pragma unroll
    for (int n = 0; n < 4; ++n)
#pragma unroll
        for (int v = 0; v < 8; ++v) {
            int row = rowBlk + wave * 16 + v + 8 * half;
            int col = colBlk + n * 16 + r;
            out[row * ldo + col] = acc[n][v];
        }
}

// ---- Kernel 4: GRKAN tail: gate/skip/layernorm/softmax ---------------------
__global__ __launch_bounds__(256) void grkan_final_kernel(
    const float* __restrict__ gatein, const float* __restrict__ skipb,
    const float* __restrict__ gamma, const float* __restrict__ beta,
    float* __restrict__ weights) {
    __shared__ float red[256];
    int b = blockIdx.x, u = threadIdx.x;
    float uu  = gatein[b * 512 + u];
    float vv  = gatein[b * 512 + 256 + u];
    float res = uu * sigmoidf_(vv) + skipb[b * U_ + u];

    red[u] = res; __syncthreads();
    for (int s = 128; s > 0; s >>= 1) { if (u < s) red[u] += red[u + s]; __syncthreads(); }
    float mu = red[0] * (1.0f / U_); __syncthreads();

    float d = res - mu;
    red[u] = d * d; __syncthreads();
    for (int s = 128; s > 0; s >>= 1) { if (u < s) red[u] += red[u + s]; __syncthreads(); }
    float var = red[0] * (1.0f / U_); __syncthreads();

    float nrm = d * rsqrtf(var + 1e-5f) * gamma[u] + beta[u];

    red[u] = nrm; __syncthreads();
    for (int s = 128; s > 0; s >>= 1) { if (u < s) red[u] = fmaxf(red[u], red[u + s]); __syncthreads(); }
    float mx = red[0]; __syncthreads();

    float e = __expf(nrm - mx);
    red[u] = e; __syncthreads();
    for (int s = 128; s > 0; s >>= 1) { if (u < s) red[u] += red[u + s]; __syncthreads(); }
    float denom = red[0];

    weights[b * U_ + u] = e / denom;
}

// ---- Kernel 5: per-step fused GEMM: [x_t | h] @ [wg | wsub] ----------------
__global__ __launch_bounds__(128) void step1_kernel(
    const float* __restrict__ xin, int t, const float* __restrict__ h,
    const float* __restrict__ wg, const float* __restrict__ bg,
    const float* __restrict__ wsub, float* __restrict__ zpre,
    float* __restrict__ ybuf) {
    __shared__ __bf16 As[TM * BK];
    __shared__ __bf16 Bt[TN * BK];
    int tid = threadIdx.x;
    int wave = tid >> 5, lane = tid & 31;
    int half = lane >> 4, r = lane & 15;
    int rowBlk = blockIdx.x * TM;
    int colBlk = blockIdx.y * TN;
    const int K = F_ + U_;  // 320
    int arr = tid >> 1, akh = (tid & 1) * 16;
    int bc = tid & 63, bkh = (tid >> 6) * 16;
    int colg_b = colBlk + bc;

    floatx8 acc[4];
#pragma unroll
    for (int n = 0; n < 4; ++n)
#pragma unroll
        for (int e = 0; e < 8; ++e) acc[n][e] = 0.0f;

    for (int k0 = 0; k0 < K; k0 += BK) {
        {
            int kc = k0 + akh;
            int rg = rowBlk + arr;
            const float* src = (kc < F_) ? &xin[(rg * T_ + t) * F_ + kc]
                                         : &h[rg * U_ + (kc - F_)];
            float v[16];
#pragma unroll
            for (int q = 0; q < 16; q += 4) {
                float4 f4 = *(const float4*)(src + q);
                v[q + 0] = f4.x; v[q + 1] = f4.y; v[q + 2] = f4.z; v[q + 3] = f4.w;
            }
            store8(&As[arr * BK + akh], v);
            store8(&As[arr * BK + akh + 8], v + 8);
        }
        {
            float v[16];
            if (colg_b < 768) {
#pragma unroll
                for (int q = 0; q < 16; ++q)
                    v[q] = wg[(k0 + bkh + q) * 768 + colg_b];
            } else {
#pragma unroll
                for (int q = 0; q < 16; ++q)
                    v[q] = wsub[(k0 + bkh + q) * U_ + (colg_b - 768)];
            }
            store8(&Bt[bc * BK + bkh], v);
            store8(&Bt[bc * BK + bkh + 8], v + 8);
        }
        __syncthreads();
        mma_step(As, Bt, wave, acc);
        __syncthreads();
    }
#pragma unroll
    for (int n = 0; n < 4; ++n)
#pragma unroll
        for (int v = 0; v < 8; ++v) {
            int row  = rowBlk + wave * 16 + v + 8 * half;
            int colg = colBlk + n * 16 + r;
            float val = acc[n][v];
            if (colg < 768) zpre[row * 768 + colg] = val + bg[colg];
            else            ybuf[row * U_ + (colg - 768)] = val;
        }
}

// ---- Host-side launch ------------------------------------------------------
extern "C" void kernel_launch(void* const* d_in, const int* in_sizes, int n_in,
                              void* d_out, int out_size, void* d_ws, size_t ws_size,
                              hipStream_t stream) {
    (void)in_sizes; (void)n_in; (void)out_size; (void)ws_size;
    const float* inputs      = (const float*)d_in[0];
    const float* time_kernel = (const float*)d_in[1];
    const float* hidden_init = (const float*)d_in[2];
    const float* g_wb1       = (const float*)d_in[3];
    const float* g_ws1       = (const float*)d_in[4];
    const float* g_wb2       = (const float*)d_in[5];
    const float* g_ws2       = (const float*)d_in[6];
    const float* g_gate      = (const float*)d_in[7];
    const float* g_skip      = (const float*)d_in[8];
    const float* g_gamma     = (const float*)d_in[9];
    const float* g_beta      = (const float*)d_in[10];
    const float* wg          = (const float*)d_in[11];
    const float* bg          = (const float*)d_in[12];
    const float* wsub        = (const float*)d_in[13];
    const float* c_wb        = (const float*)d_in[14];
    const float* c_ws        = (const float*)d_in[15];
    float* ys = (float*)d_out;

    float* w       = (float*)d_ws;
    float* sig     = w;                  w += B_ * SIG_;
    float* h1      = w;                  w += B_ * U_;
    float* h2      = w;                  w += B_ * U_;
    float* gatein  = w;                  w += B_ * 2 * U_;
    float* skipb   = w;                  w += B_ * U_;
    float* wts     = w;                  w += B_ * U_;
    float* hstate  = w;                  w += B_ * U_;
    float* zpre    = w;                  w += B_ * 3 * U_;
    float* ybuf    = w;                  w += B_ * U_;

    sig_kernel<<<dim3(B_), dim3(256), 0, stream>>>(inputs, time_kernel, sig);

    kan_kernel<<<dim3(4, 4), dim3(128), 0, stream>>>(sig, SIG_, g_wb1, g_ws1, h1, 1,
                                                     nullptr, nullptr, nullptr, nullptr, 0);
    kan_kernel<<<dim3(4, 4), dim3(128), 0, stream>>>(h1, U_, g_wb2, g_ws2, h2, 0,
                                                     nullptr, nullptr, nullptr, nullptr, 0);

    gemm_kernel<<<dim3(4, 8), dim3(128), 0, stream>>>(h2, U_, U_, g_gate, 2 * U_,
                                                      gatein, 2 * U_);
    gemm_kernel<<<dim3(4, 4), dim3(128), 0, stream>>>(sig, SIG_, SIG_, g_skip, U_,
                                                      skipb, U_);
    grkan_final_kernel<<<dim3(B_), dim3(256), 0, stream>>>(gatein, skipb, g_gamma,
                                                           g_beta, wts);

    gemm_kernel<<<dim3(4, 4), dim3(128), 0, stream>>>(inputs, T_ * F_, F_, hidden_init,
                                                      U_, hstate, U_);

    for (int t = 0; t < T_; ++t) {
        step1_kernel<<<dim3(4, 16), dim3(128), 0, stream>>>(inputs, t, hstate, wg, bg,
                                                            wsub, zpre, ybuf);
        kan_kernel<<<dim3(4, 4), dim3(128), 0, stream>>>(ybuf, U_, c_wb, c_ws,
                                                         nullptr, 0, zpre, wts,
                                                         hstate, ys, t);
    }
}
